// VariableExpander_4569845202998
// MI455X (gfx1250) — matrix operand order repaired and checked by reading, not compiled, not run on hardware
//
#include <hip/hip_runtime.h>

#define C       1536
#define MA      1536
#define KTILES  48            // 1536 / 32
#define NTILES  96            // 1536 / 16
#define MATEL   (C * C)       // 2359296 elements per matrix
#define NWG     24            // grid barrier participants
#define TPB     128           // 4 waves per block
#define WPB     4

typedef __attribute__((ext_vector_type(16))) __bf16 v16bf;
typedef __attribute__((ext_vector_type(8)))  __bf16 v8bf;
typedef __attribute__((ext_vector_type(8)))  float  v8f;

// ---- workspace layout (bytes) ----
#define OFF_W    0ull
#define W_BYTES  (6ull * (unsigned long long)MATEL * 2ull)     // 28,311,552 B bf16 weights
#define OFF_BIAS (OFF_W + W_BYTES)                             // float [3][C]
#define OFF_CUR  (OFF_BIAS + 3ull * C * 4ull)                  // ushort [2][C]
#define OFF_HID  (OFF_CUR + 2ull * C * 2ull)                   // ushort [2][3][C]
#define OFF_CNT  (OFF_HID + 6ull * C * 2ull)                   // int
#define OFF_CV   (OFF_CNT + 4ull)                              // int

__device__ __forceinline__ unsigned short f2bf(float f) {
    unsigned u = __float_as_uint(f);
    u += 0x7FFFu + ((u >> 16) & 1u);          // round-to-nearest-even
    return (unsigned short)(u >> 16);
}

// A fragment for a 1x32 vector chunk in matrix row 0, built from two
// UNCONDITIONAL 16B LDS loads: active lanes (0,16) read the real vector,
// all other lanes read a zeroed LDS pad -> no exec-mask juggling, no
// per-iteration zero re-materialization.
__device__ __forceinline__ v16bf load_a2(const unsigned short* p0,
                                         const unsigned short* p1, int ko) {
    v8bf lo = *(const v8bf*)(p0 + ko);
    v8bf hi = *(const v8bf*)(p1 + ko);
    return __builtin_shufflevector(lo, hi, 0,1,2,3,4,5,6,7,8,9,10,11,12,13,14,15);
}

__device__ __forceinline__ void gridbar(int* cnt, int gen) {
    __syncthreads();
    if (threadIdx.x == 0) {
        __threadfence();
        __hip_atomic_fetch_add(cnt, 1, __ATOMIC_ACQ_REL, __HIP_MEMORY_SCOPE_AGENT);
        while (__hip_atomic_load(cnt, __ATOMIC_ACQUIRE, __HIP_MEMORY_SCOPE_AGENT) < gen * NWG)
            __builtin_amdgcn_s_sleep(1);
        __threadfence();
    }
    __syncthreads();
}

// ---------------- kernels ----------------

__global__ void ve_init(unsigned short* curb, unsigned short* hidb, int* cnt, int* cval) {
    const int tid = threadIdx.x;
    for (int i = tid; i < 2 * C; i += blockDim.x) curb[i] = 0;
    for (int i = tid; i < 6 * C; i += blockDim.x) hidb[i] = 0;
    if (tid == 0) { *cnt = 0; *cval = 0; }
}

// fp32 -> bf16 weight conversion with WMMA-fragment swizzle:
// block (tile,kbi) is 1 KB contiguous; lane ll stores row tile*16+(ll&15),
// cols kbi*32 + (ll>>4)*16 + j  (j = 0..15) -> exactly the per-lane B fragment.
__global__ void ve_convert(const float* __restrict__ Wih, const float* __restrict__ Whh,
                           unsigned short* __restrict__ wf) {
    const int gid = blockIdx.x * blockDim.x + threadIdx.x;    // [0, 6*MATEL)
    const int mat = gid / MATEL;
    const int r   = gid % MATEL;
    const int tile = r / (KTILES * 512);
    const int r2   = r % (KTILES * 512);
    const int kbi  = r2 >> 9;
    const int r3   = r2 & 511;
    const int ll   = r3 >> 4;
    const int j    = r3 & 15;
    const int row  = tile * 16 + (ll & 15);
    const int col  = kbi * 32 + ((ll >> 4) << 4) + j;
    const float* src = (mat < 3) ? (Wih + (size_t)mat * MATEL)
                                 : (Whh + (size_t)(mat - 3) * MATEL);
    wf[gid] = f2bf(src[(size_t)row * C + col]);
}

// Head MLP (one-time, tiny): length + step count c, combined biases, initial hid.
__global__ void __launch_bounds__(256) ve_head(
        const float* __restrict__ x,
        const float* __restrict__ lw0, const float* __restrict__ lb0,
        const float* __restrict__ lw1, const float* __restrict__ lb1,
        const float* __restrict__ lw2, const float* __restrict__ lb2,
        const float* __restrict__ bih, const float* __restrict__ bhh,
        const int* __restrict__ ma_p,
        float* __restrict__ bias, unsigned short* __restrict__ hidb,
        int* __restrict__ cval, float* __restrict__ out_len) {
    __shared__ float xs[C], h0[C], h1[C], red[256];
    const int tid = threadIdx.x;
    for (int i = tid; i < C; i += 256) xs[i] = x[i];
    __syncthreads();
    for (int jj = tid; jj < C; jj += 256) {
        const float* w = lw0 + (size_t)jj * C;
        float s = lb0[jj];
        for (int k = 0; k < C; ++k) s = fmaf(w[k], xs[k], s);
        h0[jj] = (s >= 0.f) ? s : 0.2f * s;
    }
    __syncthreads();
    for (int jj = tid; jj < C; jj += 256) {
        const float* w = lw1 + (size_t)jj * C;
        float s = lb1[jj];
        for (int k = 0; k < C; ++k) s = fmaf(w[k], h0[k], s);
        h1[jj] = (s >= 0.f) ? s : 0.2f * s;
    }
    __syncthreads();
    float p = 0.f;
    for (int k = tid; k < C; k += 256) p = fmaf(lw2[k], h1[k], p);
    red[tid] = p;
    __syncthreads();
    for (int s = 128; s > 0; s >>= 1) {
        if (tid < s) red[tid] += red[tid + s];
        __syncthreads();
    }
    if (tid == 0) {
        const float l   = red[0] + lb2[0];
        const float len = fminf(fabsf(l), 0.9999f);
        *out_len = len;
        *cval = (int)floorf(len * (float)(*ma_p)) + 1;
    }
    for (int i = tid; i < 3 * C; i += 256) bias[i] = bih[i] + bhh[i];
    // first read of hid[0] is at g==0 from slot 1 -> place bf16(x) there
    for (int n = tid; n < C; n += 256) hidb[(1 * 3 + 0) * C + n] = f2bf(xs[n]);
}

// Persistent RNN: 24 WGs x 4 waves = 96 waves, wave w owns N-tile w (16 outputs).
__global__ void __launch_bounds__(TPB) ve_rnn(
        const unsigned short* __restrict__ wf, const float* __restrict__ bias,
        unsigned short* __restrict__ curb, unsigned short* __restrict__ hidb,
        int* cnt, const int* __restrict__ cval, float* __restrict__ out) {
    __shared__ __align__(32) unsigned short cur_s[C];
    __shared__ __align__(32) unsigned short hid_s[C];
    __shared__ __align__(32) unsigned short zpad[16];   // stays all-zero
    const int tid  = threadIdx.x;
    const int lane = tid & 31;
    const int wave = tid >> 5;
    const int tile = blockIdx.x * WPB + wave;
    const int cfin = *cval;

    if (tid < 16) zpad[tid] = 0;
    __syncthreads();

    // Loop-invariant per-lane A-fragment base pointers.
    const bool act  = ((lane & 15) == 0);
    const int  sel  = (lane >> 4) << 4;                 // 0 or 16
    const unsigned kmask = act ? 0xFFFFFFFFu : 0u;      // gates the kb offset
    const unsigned short* c0 = act ? (cur_s + sel)      : zpad;
    const unsigned short* c1 = act ? (cur_s + sel + 8)  : zpad;
    const unsigned short* h0 = act ? (hid_s + sel)      : zpad;
    const unsigned short* h1 = act ? (hid_s + sel + 8)  : zpad;

    int gen = 0;
    for (int t = 0; t < MA; ++t) {
        for (int l = 0; l < 3; ++l) {
            const int g  = t * 3 + l;
            const int sw = g & 1;            // write slot
            const int sr = sw ^ 1;           // read slot (written at g-1 / g-3)
            ++gen;
            gridbar(cnt, gen);

            // stage activation vectors into LDS (768 dword copies each)
            {
                const unsigned* cg = (const unsigned*)(curb + sr * C);
                const unsigned* hg = (const unsigned*)(hidb + (size_t)(sr * 3 + l) * C);
                unsigned* cs = (unsigned*)cur_s;
                unsigned* hs = (unsigned*)hid_s;
                for (int i = tid; i < C / 2; i += TPB) { cs[i] = cg[i]; hs[i] = hg[i]; }
            }
            __syncthreads();

            const unsigned short* wih = wf + (size_t)l * MATEL + (size_t)tile * (KTILES * 512);
            const unsigned short* whh = wf + (size_t)(3 + l) * MATEL + (size_t)tile * (KTILES * 512);

            v8f acc = {};
            for (int kbi = 0; kbi < KTILES; ++kbi) {
                const int ko = (int)((unsigned)(kbi * 32) & kmask);
                const v16bf ac = load_a2(c0, c1, ko);
                const v16bf ah = load_a2(h0, h1, ko);
                const v16bf bi = *(const v16bf*)(wih + kbi * 512 + lane * 16);
                const v16bf bh = *(const v16bf*)(whh + kbi * 512 + lane * 16);
                __builtin_prefetch(wih + (kbi + 1) * 512 + lane * 16, 0, 1);
                __builtin_prefetch(whh + (kbi + 1) * 512 + lane * 16, 0, 1);
                acc = __builtin_amdgcn_wmma_f32_16x16x32_bf16(false, ac, false, bi,
                                                              (short)0, acc, false, false);
                acc = __builtin_amdgcn_wmma_f32_16x16x32_bf16(false, ah, false, bh,
                                                              (short)0, acc, false, false);
            }

            if (lane < 16) {                  // D row 0: VGPR0, lanes 0..15 = N
                const int n = tile * 16 + lane;
                const float y  = acc[0] + bias[l * C + n];
                const float nh = fmaxf(y, 0.f);
                const unsigned short nb = f2bf(nh);
                hidb[(size_t)(sw * 3 + l) * C + n] = nb;
                curb[sw * C + n] = nb;
                if (l == 2) out[(size_t)t * C + n] = (t < cfin) ? nh : 0.f;
            }
        }
    }
}

extern "C" void kernel_launch(void* const* d_in, const int* in_sizes, int n_in,
                              void* d_out, int out_size, void* d_ws, size_t ws_size,
                              hipStream_t stream) {
    const float* x   = (const float*)d_in[0];
    const float* lw0 = (const float*)d_in[1];
    const float* lb0 = (const float*)d_in[2];
    const float* lw1 = (const float*)d_in[3];
    const float* lb1 = (const float*)d_in[4];
    const float* lw2 = (const float*)d_in[5];
    const float* lb2 = (const float*)d_in[6];
    const float* Wih = (const float*)d_in[7];
    const float* bih = (const float*)d_in[8];
    const float* Whh = (const float*)d_in[9];
    const float* bhh = (const float*)d_in[10];
    const int*   map = (const int*)d_in[11];
    float* out = (float*)d_out;

    char* ws = (char*)d_ws;
    unsigned short* wf   = (unsigned short*)(ws + OFF_W);
    float*          bias = (float*)(ws + OFF_BIAS);
    unsigned short* curb = (unsigned short*)(ws + OFF_CUR);
    unsigned short* hidb = (unsigned short*)(ws + OFF_HID);
    int*            cnt  = (int*)(ws + OFF_CNT);
    int*            cval = (int*)(ws + OFF_CV);

    ve_init<<<1, 256, 0, stream>>>(curb, hidb, cnt, cval);
    ve_convert<<<(6 * MATEL) / 256, 256, 0, stream>>>(Wih, Whh, wf);
    ve_head<<<1, 256, 0, stream>>>(x, lw0, lb0, lw1, lb1, lw2, lb2, bih, bhh, map,
                                   bias, hidb, cval, out + (size_t)MA * C);
    ve_rnn<<<NWG, TPB, 0, stream>>>(wf, bias, curb, hidb, cnt, cval, out);
    (void)in_sizes; (void)n_in; (void)out_size; (void)ws_size;
}
